// CLGF_GNNDrug_67877663146715
// MI455X (gfx1250) — matrix operand order repaired and checked by reading, not compiled, not run on hardware
//
#include <hip/hip_runtime.h>
#include <math.h>

// CLGF_GNNDrug: GCN(2) + GIN(3) + concat + segment_max for MI455X (gfx1250).
// GEMMs via v_wmma_f32_16x16x32_f16 (wave32, 16x16 tiles, f32 accum),
// A tiles staged through LDS (ds_load_b128), B fragments register-resident.
// Epilogue fully templated: no branches, immediate-offset stores.
// Edge scatters / pooling via global f32 atomics (memory-bound part).

typedef __attribute__((ext_vector_type(16))) _Float16 v16h;
typedef __attribute__((ext_vector_type(8)))  _Float16 v8h;
typedef __attribute__((ext_vector_type(8)))  float    v8f;

#define DD 128

// ---------------- elementwise / utility ----------------
__global__ void k_fill(float* p, float v, int n) {
    int i = blockIdx.x * blockDim.x + threadIdx.x;
    if (i < n) p[i] = v;
}
__global__ void k_copy(float* dst, const float* src, int n) {
    int i = blockIdx.x * blockDim.x + threadIdx.x;
    if (i < n) dst[i] = src[i];
}
// out[r,c] = (r<rin && c<cin) ? in[r,c] : 0   (f32 -> f16, zero pad)
__global__ void k_cvt_f16(const float* __restrict__ in, _Float16* __restrict__ out,
                          int rin, int cin, int rout, int cout) {
    int i = blockIdx.x * blockDim.x + threadIdx.x;
    if (i >= rout * cout) return;
    int r = i / cout, c = i - r * cout;
    float v = (r < rin && c < cin) ? in[(size_t)r * cin + c] : 0.0f;
    out[i] = (_Float16)v;
}

// ---------------- degree / norm ----------------
__global__ void k_deg_count(const int* __restrict__ dst, float* deg, int E) {
    int i = blockIdx.x * blockDim.x + threadIdx.x;
    if (i < E) atomicAdd(&deg[dst[i]], 1.0f);
}
__global__ void k_dinv(const float* deg, float* dinv, int n) {
    int i = blockIdx.x * blockDim.x + threadIdx.x;
    if (i < n) dinv[i] = rsqrtf(deg[i]);
}

// ---------------- GCN aggregation ----------------
__global__ void k_gcn_selfinit(const float* __restrict__ h, const float* __restrict__ dinv,
                               float* __restrict__ agg, int n) {
    int i = blockIdx.x * blockDim.x + threadIdx.x;
    if (i >= n * DD) return;
    float di = dinv[i >> 7];
    agg[i] = h[i] * di * di;
}
__global__ void k_gcn_scatter(const float* __restrict__ h, const int* __restrict__ src,
                              const int* __restrict__ dst, const float* __restrict__ dinv,
                              float* __restrict__ agg, int E) {
    int i = blockIdx.x * blockDim.x + threadIdx.x;   // E * 32 threads (4 cols each)
    int e = i >> 5;
    if (e >= E) return;
    int c = (i & 31) * 4;
    int s = src[e], d = dst[e];
    float nrm = dinv[s] * dinv[d];
    const float4 hv = *(const float4*)(h + (size_t)s * DD + c);
    float* o = agg + (size_t)d * DD + c;
    atomicAdd(o + 0, hv.x * nrm);
    atomicAdd(o + 1, hv.y * nrm);
    atomicAdd(o + 2, hv.z * nrm);
    atomicAdd(o + 3, hv.w * nrm);
}

// out = relu(in + bias); optional f32 and f16 destinations
__global__ void k_bias_relu(const float* __restrict__ in, const float* __restrict__ bias,
                            float* __restrict__ out32, _Float16* __restrict__ out16, int n) {
    int i = blockIdx.x * blockDim.x + threadIdx.x;
    if (i >= n * DD) return;
    float v = fmaxf(in[i] + bias[i & 127], 0.0f);
    if (out32) out32[i] = v;
    if (out16) out16[i] = (_Float16)v;
}

// ---------------- GIN aggregation: acc[dst] += x[src] ----------------
__global__ void k_gin_scatter(const float* __restrict__ x, const int* __restrict__ src,
                              const int* __restrict__ dst, float* __restrict__ acc, int C) {
    int e = blockIdx.x;
    int c = threadIdx.x;
    if (c >= C) return;
    int s = src[e], d = dst[e];
    atomicAdd(&acc[(size_t)d * C + c], x[(size_t)s * C + c]);
}

// ---------------- batch norm (training-mode, biased var) ----------------
__global__ void k_bn_stats(const float* __restrict__ v, float* sums, float* sumsq, int n) {
    int c = threadIdx.x;   // blockDim.x == 128
    float s = 0.0f, q = 0.0f;
    for (int r = blockIdx.x; r < n; r += gridDim.x) {
        float x = v[(size_t)r * DD + c];
        s += x; q += x * x;
    }
    atomicAdd(&sums[c], s);
    atomicAdd(&sumsq[c], q);
}
__global__ void k_bn_apply(const float* __restrict__ v, const float* __restrict__ sums,
                           const float* __restrict__ sumsq, const float* __restrict__ g,
                           const float* __restrict__ be, float* __restrict__ out, int n) {
    int i = blockIdx.x * blockDim.x + threadIdx.x;
    if (i >= n * DD) return;
    int c = i & 127;
    float inv_n = 1.0f / (float)n;
    float m = sums[c] * inv_n;
    float var = sumsq[c] * inv_n - m * m;
    out[i] = (v[i] - m) * rsqrtf(var + 1e-5f) * g[c] + be[c];
}

// ---------------- pooling: segment_max via sign-split int atomics ----------------
__device__ __forceinline__ void amaxf(float* a, float v) {
    if (v >= 0.0f) {
        atomicMax((int*)a, __float_as_int(v == 0.0f ? 0.0f : v));  // map -0 -> +0
    } else {
        atomicMin((unsigned int*)a, (unsigned int)__float_as_int(v));
    }
}
__global__ void k_pool_init(float* out, int n) {
    int i = blockIdx.x * blockDim.x + threadIdx.x;
    if (i < n) out[i] = -INFINITY;
}
__global__ void k_pool(const float* __restrict__ xg1, const float* __restrict__ xg2,
                       const float* __restrict__ h1, const float* __restrict__ h2,
                       const float* __restrict__ h3, const int* __restrict__ batch,
                       float* __restrict__ out, int n) {
    int i = blockIdx.x * blockDim.x + threadIdx.x;
    if (i >= n * DD) return;
    int node = i >> 7, c = i & 127;
    float* o = out + (size_t)batch[node] * (9 * DD);
    float a = xg1[i], b = xg2[i], p = h1[i], q = h2[i], r = h3[i];
    amaxf(o + 0 * DD + c, a);
    amaxf(o + 1 * DD + c, b);
    amaxf(o + 2 * DD + c, a + b);
    amaxf(o + 3 * DD + c, a * b);
    amaxf(o + 4 * DD + c, p);
    amaxf(o + 5 * DD + c, q);
    amaxf(o + 6 * DD + c, r);
    amaxf(o + 7 * DD + c, p + q + r);
    amaxf(o + 8 * DD + c, p * q * r);
}

// ---------------- WMMA GEMM:  C[N x 128] = act(A[N x KP] @ B[KP x 128] + bias) ----------------
// Block = 256 threads = 8 wave32s; wave w owns column tile [16w, 16w+16).
// Per-wave B fragments live in VGPRs for the whole kernel (reused across row tiles).
// A tile (16 x KP f16) staged through LDS with 128-bit loads.
// Layouts per CDNA5 ISA:
//   A frag : lane l -> row (l&15); elems 0..7 = K kb..kb+7, 8..15 = K kb+16..kb+23, kb=(l<16)?0:8
//   B frag : lane l -> col (l&15); elems 0..7 = K kb..kb+7, 8..15 = K kb+16..kb+23, kb=(l<16)?0:8
//   C/D    : elem r -> row r + 8*(l>>4), col (l&15)
template <int KP, bool RELU, bool HASBIAS, bool OUT16>
__global__ __launch_bounds__(256) void k_gemm(const _Float16* __restrict__ A,
                                              const _Float16* __restrict__ B,
                                              const float* __restrict__ bias,
                                              void* __restrict__ outp,
                                              int ntiles) {
    constexpr int NK = KP / 32;
    __shared__ _Float16 As[16 * KP];

    const int lane = threadIdx.x & 31;
    const int wave = threadIdx.x >> 5;           // 0..7 : column tile
    const int ncol = wave * 16 + (lane & 15);    // output column owned by this lane
    const int kb_b = (lane < 16) ? 0 : 8;

    v16h bfr[NK];
#pragma unroll
    for (int s = 0; s < NK; ++s) {
#pragma unroll
        for (int e = 0; e < 16; ++e)
            bfr[s][e] = B[(size_t)(s * 32 + kb_b + e + ((e >> 3) << 3)) * DD + ncol];
    }
    float bv = 0.0f;
    if (HASBIAS) bv = bias[ncol];

    const int m_a  = lane & 15;
    const int kb_a = (lane < 16) ? 0 : 8;
    const int m0   = (lane < 16) ? 0 : 8;

    for (int t = blockIdx.x; t < ntiles; t += gridDim.x) {
        const size_t r0 = (size_t)t * 16;
        // stage A tile (16 rows x KP halves) into LDS, one uint4 (8 halves) per thread
        {
            const int nchunk = 16 * (KP / 8);    // 256 (KP=128) or 192 (KP=96)
            int tid = threadIdx.x;
            if (tid < nchunk) {
                int row = tid / (KP / 8);
                int ch  = tid - row * (KP / 8);
                *(uint4*)(&As[row * KP + ch * 8]) =
                    *(const uint4*)(A + (r0 + row) * KP + ch * 8);
            }
        }
        __syncthreads();

        v8f acc;
#pragma unroll
        for (int r = 0; r < 8; ++r) acc[r] = bv;

#pragma unroll
        for (int s = 0; s < NK; ++s) {
            const v8h lo = *(const v8h*)(&As[m_a * KP + s * 32 + kb_a]);
            const v8h hi = *(const v8h*)(&As[m_a * KP + s * 32 + kb_a + 16]);
            v16h af;
#pragma unroll
            for (int e = 0; e < 8; ++e) { af[e] = lo[e]; af[8 + e] = hi[e]; }
            acc = __builtin_amdgcn_wmma_f32_16x16x32_f16(
                false, af, false, bfr[s], (short)0, acc, false, false);
        }

        // store 8 rows (stride DD) from one per-lane base pointer -> imm-offset stores
        if (OUT16) {
            _Float16* p = (_Float16*)outp + (r0 + m0) * DD + ncol;
#pragma unroll
            for (int r = 0; r < 8; ++r) {
                float v = acc[r];
                if (RELU) v = fmaxf(v, 0.0f);
                p[r * DD] = (_Float16)v;
            }
        } else {
            float* p = (float*)outp + (r0 + m0) * DD + ncol;
#pragma unroll
            for (int r = 0; r < 8; ++r) {
                float v = acc[r];
                if (RELU) v = fmaxf(v, 0.0f);
                p[r * DD] = v;
            }
        }
        __syncthreads();
    }
}

// ---------------- driver ----------------
extern "C" void kernel_launch(void* const* d_in, const int* in_sizes, int n_in,
                              void* d_out, int out_size, void* d_ws, size_t ws_size,
                              hipStream_t stream) {
    const int F_IN = 77;
    const int N  = in_sizes[0] / F_IN;
    const int E  = in_sizes[1] / 2;
    const int NG = out_size / (9 * DD);
    const int ntiles = N / 16;                   // N = 100000 is a multiple of 16

    const float* x     = (const float*)d_in[0];
    const int*   ei    = (const int*)d_in[1];
    const int*   src   = ei;
    const int*   dst   = ei + E;
    const int*   batch = (const int*)d_in[2];
    const float* W1 = (const float*)d_in[3];
    const float* b1 = (const float*)d_in[4];
    const float* W2 = (const float*)d_in[5];
    const float* b2 = (const float*)d_in[6];
    const float *A_[3], *a_[3], *B_[3], *c_[3], *g_[3], *be_[3];
    for (int i = 0; i < 3; ++i) {
        A_[i]  = (const float*)d_in[7 + 6 * i];
        a_[i]  = (const float*)d_in[8 + 6 * i];
        B_[i]  = (const float*)d_in[9 + 6 * i];
        c_[i]  = (const float*)d_in[10 + 6 * i];
        g_[i]  = (const float*)d_in[11 + 6 * i];
        be_[i] = (const float*)d_in[12 + 6 * i];
    }
    float* out = (float*)d_out;

    // ---- workspace carve-out (256B aligned) ----
    char* w = (char*)d_ws;
    size_t off = 0;
    auto take = [&](size_t bytes) -> void* {
        void* p = w + off;
        off += (bytes + 255) & ~(size_t)255;
        return p;
    };
    const size_t nd4 = (size_t)N * DD * 4;
    float* XG1 = (float*)take(nd4);
    float* XG2 = (float*)take(nd4);
    float* H1  = (float*)take(nd4);
    float* H2  = (float*)take(nd4);
    float* H3  = (float*)take(nd4);
    float* T0  = (float*)take(nd4);
    float* T1  = (float*)take(nd4);
    _Float16* Xh = (_Float16*)take((size_t)N * 96 * 2);
    _Float16* Fh = (_Float16*)take((size_t)N * DD * 2);
    _Float16* Gh = (_Float16*)take((size_t)N * DD * 2);
    float* deg  = (float*)take((size_t)N * 4);
    float* dinv = (float*)take((size_t)N * 4);
    _Float16* W1h = (_Float16*)take(96 * DD * 2);
    _Float16* W2h = (_Float16*)take(DD * DD * 2);
    _Float16* Ah[3]; _Float16* Bh[3];
    Ah[0] = (_Float16*)take(96 * DD * 2);
    Ah[1] = (_Float16*)take(DD * DD * 2);
    Ah[2] = (_Float16*)take(DD * DD * 2);
    for (int i = 0; i < 3; ++i) Bh[i] = (_Float16*)take(DD * DD * 2);
    float* bns = (float*)take(DD * 4);
    float* bnq = (float*)take(DD * 4);
    (void)ws_size; (void)n_in;

    const int TPB = 256;
    auto blocks = [&](long n) { return (unsigned)((n + TPB - 1) / TPB); };
    const unsigned GB = 1280;   // grid-stride GEMM blocks

    // ---- weight conversions to f16 (K padded to mult of 32) ----
    k_cvt_f16<<<blocks(96 * DD), TPB, 0, stream>>>(W1, W1h, F_IN, DD, 96, DD);
    k_cvt_f16<<<blocks(DD * DD), TPB, 0, stream>>>(W2, W2h, DD, DD, DD, DD);
    k_cvt_f16<<<blocks(96 * DD), TPB, 0, stream>>>(A_[0], Ah[0], F_IN, DD, 96, DD);
    k_cvt_f16<<<blocks(DD * DD), TPB, 0, stream>>>(A_[1], Ah[1], DD, DD, DD, DD);
    k_cvt_f16<<<blocks(DD * DD), TPB, 0, stream>>>(A_[2], Ah[2], DD, DD, DD, DD);
    for (int i = 0; i < 3; ++i)
        k_cvt_f16<<<blocks(DD * DD), TPB, 0, stream>>>(B_[i], Bh[i], DD, DD, DD, DD);

    // ---- degrees / symmetric norm ----
    k_fill<<<blocks(N), TPB, 0, stream>>>(deg, 1.0f, N);
    k_deg_count<<<blocks(E), TPB, 0, stream>>>(dst, deg, E);
    k_dinv<<<blocks(N), TPB, 0, stream>>>(deg, dinv, N);

    // ---- x -> f16 (pad 77 -> 96) ----
    k_cvt_f16<<<blocks((long)N * 96), TPB, 0, stream>>>(x, Xh, N, F_IN, N, 96);

    // ---- GCN layer 1 ----
    k_gemm<96, false, false, false><<<GB, TPB, 0, stream>>>(Xh, W1h, nullptr, T0, ntiles);
    k_gcn_selfinit<<<blocks((long)N * DD), TPB, 0, stream>>>(T0, dinv, T1, N);
    k_gcn_scatter<<<blocks((long)E * 32), TPB, 0, stream>>>(T0, src, dst, dinv, T1, E);
    k_bias_relu<<<blocks((long)N * DD), TPB, 0, stream>>>(T1, b1, XG1, Fh, N);

    // ---- GCN layer 2 ----
    k_gemm<128, false, false, false><<<GB, TPB, 0, stream>>>(Fh, W2h, nullptr, T0, ntiles);
    k_gcn_selfinit<<<blocks((long)N * DD), TPB, 0, stream>>>(T0, dinv, T1, N);
    k_gcn_scatter<<<blocks((long)E * 32), TPB, 0, stream>>>(T0, src, dst, dinv, T1, E);
    k_bias_relu<<<blocks((long)N * DD), TPB, 0, stream>>>(T1, b2, XG2, nullptr, N);

    // ---- GIN layer 0 (input x, width 77) ----
    k_copy<<<blocks((long)N * F_IN), TPB, 0, stream>>>(T0, x, N * F_IN);
    k_gin_scatter<<<E, DD, 0, stream>>>(x, src, dst, T0, F_IN);
    k_cvt_f16<<<blocks((long)N * 96), TPB, 0, stream>>>(T0, Xh, N, F_IN, N, 96);
    k_gemm<96, true, true, true><<<GB, TPB, 0, stream>>>(Xh, Ah[0], a_[0], Gh, ntiles);
    k_gemm<128, true, true, false><<<GB, TPB, 0, stream>>>(Gh, Bh[0], c_[0], T0, ntiles);
    k_fill<<<1, DD, 0, stream>>>(bns, 0.0f, DD);
    k_fill<<<1, DD, 0, stream>>>(bnq, 0.0f, DD);
    k_bn_stats<<<512, DD, 0, stream>>>(T0, bns, bnq, N);
    k_bn_apply<<<blocks((long)N * DD), TPB, 0, stream>>>(T0, bns, bnq, g_[0], be_[0], H1, N);

    // ---- GIN layers 1 & 2 (width 128) ----
    float* Hin[3] = {H1, H2, H3};
    for (int li = 1; li < 3; ++li) {
        const float* hin = Hin[li - 1];
        float* hout = Hin[li];
        k_copy<<<blocks((long)N * DD), TPB, 0, stream>>>(T0, hin, N * DD);
        k_gin_scatter<<<E, DD, 0, stream>>>(hin, src, dst, T0, DD);
        k_cvt_f16<<<blocks((long)N * DD), TPB, 0, stream>>>(T0, Fh, N, DD, N, DD);
        k_gemm<128, true, true, true><<<GB, TPB, 0, stream>>>(Fh, Ah[li], a_[li], Gh, ntiles);
        k_gemm<128, true, true, false><<<GB, TPB, 0, stream>>>(Gh, Bh[li], c_[li], T0, ntiles);
        k_fill<<<1, DD, 0, stream>>>(bns, 0.0f, DD);
        k_fill<<<1, DD, 0, stream>>>(bnq, 0.0f, DD);
        k_bn_stats<<<512, DD, 0, stream>>>(T0, bns, bnq, N);
        k_bn_apply<<<blocks((long)N * DD), TPB, 0, stream>>>(T0, bns, bnq, g_[li], be_[li], hout, N);
    }

    // ---- concat (on the fly) + segment_max pooling ----
    k_pool_init<<<blocks((long)NG * 9 * DD), TPB, 0, stream>>>(out, NG * 9 * DD);
    k_pool<<<blocks((long)N * DD), TPB, 0, stream>>>(XG1, XG2, H1, H2, H3, batch, out, N);
}
